// FastWeightProductKeyMemory_10325101379653
// MI455X (gfx1250) — matrix operand order repaired and checked by reading, not compiled, not run on hardware
//
#include <hip/hip_runtime.h>
#include <hip/hip_bf16.h>
#include <cstdint>
#include <cstddef>

// Problem constants (from reference)
#define HEADS   4
#define K_DIM   512
#define HALF    256
#define V_DIM   512
#define SUBSIZE 512
#define TOPK    8
#define QSTRIDE (HEADS * K_DIM)   // 2048 floats per token

typedef __attribute__((ext_vector_type(16))) __bf16 v16bf;
typedef __attribute__((ext_vector_type(8)))  __bf16 v8bf;
typedef __attribute__((ext_vector_type(8)))  float  v8f;

// ---------------------------------------------------------------------------
// Kernel 0: f32 -> bf16 conversion (query and keys) into workspace
// ---------------------------------------------------------------------------
__global__ void cvt_bf16_kernel(const float* __restrict__ src,
                                __bf16* __restrict__ dst, int n) {
    int i = blockIdx.x * blockDim.x + threadIdx.x;
    int stride = gridDim.x * blockDim.x;
    for (; i < n; i += stride) dst[i] = (__bf16)src[i];
}

// ---------------------------------------------------------------------------
// Kernel 1: score GEMM (bf16 WMMA, f32 accumulate) fused with per-row top-8.
// Grid: (ntok/16, 8). blockIdx.y = hp = p*HEADS + h  (matches keys layout).
// Block: 256 threads = 8 waves; wave w computes 16 tokens x 64 subkeys.
// ---------------------------------------------------------------------------
#define SSTR 516   // padded LDS row stride (floats) -> bank-conflict-free topk

__global__ __launch_bounds__(256) void score_topk_kernel(
    const __bf16* __restrict__ qb,   // [ntok][2048] bf16
    const __bf16* __restrict__ kb,   // [8*512][256] bf16
    float* __restrict__ ts_out,      // [ntok][8][TOPK]
    int*   __restrict__ ti_out) {    // [ntok][8][TOPK]
    __shared__ float sS[16 * SSTR];

    const int tileM = blockIdx.x;
    const int hp    = blockIdx.y;          // p*4 + h
    const int h     = hp & 3;
    const int p     = hp >> 2;
    const int lane  = threadIdx.x & 31;
    const int wave  = threadIdx.x >> 5;
    const int l15   = lane & 15;
    const int hi    = lane >> 4;           // lane group (0: lanes 0-15, 1: 16-31)

    // A: row = token within tile (per ISA 16-bit A layout)
    const int token = tileM * 16 + l15;
    const __bf16* qrow  = qb + (size_t)token * QSTRIDE + h * K_DIM + p * HALF;
    // B: key rows for this (half, head); contiguous along K (HALF) dim
    const __bf16* kbase = kb + (size_t)hp * SUBSIZE * HALF;

    const int n0 = wave * 64;              // this wave's subkey block

    v8f acc[4] = {};                       // four 16x16 f32 accumulators

    // A-layout K offsets: lanes 0-15 -> K 0..15, lanes 16-31 -> K 16..31
    const int akb0  = hi ? 16 : 0;
    const int akb1  = hi ? 24 : 8;
    // B K offsets (contiguous, same as A): lanes 0-15 hold K 0..15, lanes 16-31 hold K 16..31
    const int bkoff = hi ? 16 : 0;

    #pragma unroll
    for (int kk = 0; kk < HALF; kk += 32) {
        v8bf alo = *(const v8bf*)(qrow + kk + akb0);
        v8bf ahi = *(const v8bf*)(qrow + kk + akb1);
        v16bf A  = __builtin_shufflevector(alo, ahi,
                     0,1,2,3,4,5,6,7,8,9,10,11,12,13,14,15);
        #pragma unroll
        for (int t = 0; t < 4; ++t) {
            const int s = n0 + t * 16 + l15;          // subkey index (B column)
            v16bf Bm = *(const v16bf*)(kbase + (size_t)s * HALF + kk + bkoff);
            acc[t] = __builtin_amdgcn_wmma_f32_16x16x32_bf16(
                         false, A, false, Bm, (short)0, acc[t], false, false);
        }
    }

    // Spill accumulators to LDS per the C/D layout:
    // VGPR g: lanes 0-15 -> (M=g, N=lane), lanes 16-31 -> (M=g+8, N=lane-16)
    const int mbase = hi * 8;
    #pragma unroll
    for (int t = 0; t < 4; ++t) {
        const int c = n0 + t * 16 + l15;
        #pragma unroll
        for (int g = 0; g < 8; ++g)
            sS[(mbase + g) * SSTR + c] = acc[t][g];
    }
    __syncthreads();

    // Serial insertion top-8 per row; strict '<' shift preserves jax tie order.
    if (threadIdx.x < 16) {
        const int r = threadIdx.x;
        float tv[TOPK]; int tix[TOPK];
        #pragma unroll
        for (int j = 0; j < TOPK; ++j) { tv[j] = -__builtin_inff(); tix[j] = 0; }
        for (int s = 0; s < SUBSIZE; ++s) {
            float v = sS[r * SSTR + s];
            if (v > tv[TOPK - 1]) {
                int pos = TOPK - 1;
                while (pos > 0 && tv[pos - 1] < v) {
                    tv[pos] = tv[pos - 1]; tix[pos] = tix[pos - 1]; --pos;
                }
                tv[pos] = v; tix[pos] = s;
            }
        }
        const int tok = tileM * 16 + r;
        float* tso = ts_out + ((size_t)tok * 8 + hp) * TOPK;
        int*   tio = ti_out + ((size_t)tok * 8 + hp) * TOPK;
        #pragma unroll
        for (int j = 0; j < TOPK; ++j) { tso[j] = tv[j]; tio[j] = tix[j]; }
    }
}

// ---------------------------------------------------------------------------
// Kernel 2: per-token combine top-8, softmax(32), weighted float4 gather.
// Grid: ntok blocks x 128 threads (one float4 of V_DIM per thread).
// ---------------------------------------------------------------------------
__global__ __launch_bounds__(128) void combine_gather_kernel(
    const float* __restrict__ ts, const int* __restrict__ ti,
    const float* __restrict__ values, float* __restrict__ out) {
    __shared__ float sW[HEADS * TOPK];
    __shared__ int   sIdx[HEADS * TOPK];

    const int n   = blockIdx.x;
    const int tid = threadIdx.x;

    if (tid < HEADS) {
        const int hh = tid;
        const float* t1 = ts + ((size_t)n * 8 + hh) * TOPK;       // half 0
        const float* t2 = ts + ((size_t)n * 8 + 4 + hh) * TOPK;   // half 1
        const int*   i1 = ti + ((size_t)n * 8 + hh) * TOPK;
        const int*   i2 = ti + ((size_t)n * 8 + 4 + hh) * TOPK;
        float a1[TOPK], a2[TOPK];
        #pragma unroll
        for (int j = 0; j < TOPK; ++j) { a1[j] = t1[j]; a2[j] = t2[j]; }

        float tv[TOPK]; int tf[TOPK];
        #pragma unroll
        for (int j = 0; j < TOPK; ++j) { tv[j] = -__builtin_inff(); tf[j] = 0; }
        for (int f = 0; f < TOPK * TOPK; ++f) {
            float v = a1[f >> 3] + a2[f & 7];
            if (v > tv[TOPK - 1]) {
                int pos = TOPK - 1;
                while (pos > 0 && tv[pos - 1] < v) {
                    tv[pos] = tv[pos - 1]; tf[pos] = tf[pos - 1]; --pos;
                }
                tv[pos] = v; tf[pos] = f;
            }
        }
        #pragma unroll
        for (int j = 0; j < TOPK; ++j) {
            const int f = tf[j];
            sIdx[hh * TOPK + j] = i1[f >> 3] * SUBSIZE + i2[f & 7];
            sW[hh * TOPK + j]   = tv[j];     // raw scores; softmax below
        }
    }
    __syncthreads();

    if (tid == 0) {   // TEMP == 1.0
        float m = -__builtin_inff();
        for (int k = 0; k < HEADS * TOPK; ++k) m = fmaxf(m, sW[k]);
        float e[HEADS * TOPK]; float sum = 0.f;
        for (int k = 0; k < HEADS * TOPK; ++k) { e[k] = __expf(sW[k] - m); sum += e[k]; }
        const float inv = 1.f / sum;
        for (int k = 0; k < HEADS * TOPK; ++k) sW[k] = e[k] * inv;
    }
    __syncthreads();

    // Weighted embedding-bag gather: coalesced float4 reads of values rows.
    const float4* v4 = (const float4*)values;
    float4 acc = make_float4(0.f, 0.f, 0.f, 0.f);
    #pragma unroll 8
    for (int k = 0; k < HEADS * TOPK; ++k) {
        const float w = sW[k];
        const float4 x = v4[(size_t)sIdx[k] * (V_DIM / 4) + tid];
        acc.x += w * x.x; acc.y += w * x.y; acc.z += w * x.z; acc.w += w * x.w;
    }
    ((float4*)out)[(size_t)n * (V_DIM / 4) + tid] = acc;
}

// ---------------------------------------------------------------------------
// Launch
// ---------------------------------------------------------------------------
extern "C" void kernel_launch(void* const* d_in, const int* in_sizes, int n_in,
                              void* d_out, int out_size, void* d_ws, size_t ws_size,
                              hipStream_t stream) {
    const float* q      = (const float*)d_in[0];   // [ntok, 2048]
    const float* keys   = (const float*)d_in[1];   // [4096, 256]
    const float* values = (const float*)d_in[2];   // [262144, 512]
    float* out = (float*)d_out;

    const int nq   = in_sizes[0];                  // ntok * 2048
    const int nk   = in_sizes[1];                  // 4096 * 256
    const int ntok = nq / QSTRIDE;                 // 8192

    // Workspace layout
    char* ws = (char*)d_ws;
    __bf16* qb = (__bf16*)ws;                                  // nq * 2 bytes
    __bf16* kb = (__bf16*)(ws + (size_t)nq * 2);               // nk * 2 bytes
    float*  ts = (float*)(ws + (size_t)nq * 2 + (size_t)nk * 2);
    int*    ti = (int*)((char*)ts + (size_t)ntok * 8 * TOPK * sizeof(float));

    cvt_bf16_kernel<<<4096, 256, 0, stream>>>(q, qb, nq);
    cvt_bf16_kernel<<<1024, 256, 0, stream>>>(keys, kb, nk);

    dim3 gA(ntok / 16, 2 * HEADS);
    score_topk_kernel<<<gA, 256, 0, stream>>>(qb, kb, ts, ti);

    combine_gather_kernel<<<ntok, 128, 0, stream>>>(ts, ti, values, out);
}
